// SpatialILFusionModule_20212116095638
// MI455X (gfx1250) — matrix operand order repaired and checked by reading, not compiled, not run on hardware
//
#include <hip/hip_runtime.h>
#include <stdint.h>

typedef __attribute__((ext_vector_type(16))) _Float16 v16h;
typedef __attribute__((ext_vector_type(8)))  _Float16 v8h;
typedef __attribute__((ext_vector_type(8)))  float    v8f;

// ---------------------------------------------------------------------------
// CDNA5 wave32 WMMA fragment layouts (V_WMMA_F32_16X16X32_F16):
//  A 16(M)x32(K): lane l -> row m=l&15, half h=l>>4; v16h = A[m][8h..8h+7] ++ A[m][16+8h..16+8h+7]
//  B 32(K)x16(N): lane l -> col n=l&15, half h=l>>4; v16h = B[8h..8h+7][n] ++ B[16+8h..16+8h+7][n]
//     => with column-major LDS (Bs[n][k]) the fragment is two contiguous runs of 8 f16, 16 apart.
//  C/D 16x16 f32: lane l col n=l&15, VGPR j row m=(l>>4)*8+j
// ---------------------------------------------------------------------------
__device__ inline v16h frag_cat(v8h lo, v8h hi) {
  return __builtin_shufflevector(lo, hi, 0, 1, 2, 3, 4, 5, 6, 7, 8, 9, 10, 11, 12, 13, 14, 15);
}

__device__ inline v16h load_a_frag_rm(const _Float16* As, int rowBase, int lane) {
  const _Float16* p = As + (rowBase + (lane & 15)) * 32;
  int h = lane >> 4;
  v8h lo = *(const v8h*)(p + 8 * h);
  v8h hi = *(const v8h*)(p + 16 + 8 * h);
  return frag_cat(lo, hi);
}

__device__ inline v16h load_b_frag_cm(const _Float16* Bs, int colBase, int krow, int ldk, int lane) {
  const _Float16* p = Bs + (colBase + (lane & 15)) * ldk + krow + 8 * (lane >> 4);
  v8h lo = *(const v8h*)p;
  v8h hi = *(const v8h*)(p + 16);
  return frag_cat(lo, hi);
}

__device__ inline v8f wmma16(v16h a, v16h b, v8f c) {
  return __builtin_amdgcn_wmma_f32_16x16x32_f16(false, a, false, b, (short)0, c, false, false);
}

// Async global->LDS 16B copy (GLOBAL_LOAD_ASYNC_TO_LDS_B128, ASYNCcnt-tracked).
__device__ inline void async_copy16(const void* gptr, void* lptr) {
  unsigned lds = (unsigned)(uintptr_t)lptr;
  uint64_t g = (uint64_t)(uintptr_t)gptr;
  asm volatile("global_load_async_to_lds_b128 %0, %1, off" ::"v"(lds), "v"(g) : "memory");
}
__device__ inline void async_wait0() { asm volatile("s_wait_asynccnt 0x0" ::: "memory"); }

// ---------------------------------------------------------------------------
// Small utility kernels
// ---------------------------------------------------------------------------
__global__ void zero_f32_kernel(float* p, int n) {
  int i = blockIdx.x * 256 + threadIdx.x;
  if (i < n) p[i] = 0.0f;
}

__global__ void f32_to_f16_kernel(const float* __restrict__ s, _Float16* __restrict__ d, int n) {
  int i = blockIdx.x * 256 + threadIdx.x;
  if (i < n) d[i] = (_Float16)s[i];
}

__global__ void scatter_mask_kernel(const float* __restrict__ coords,
                                    float* __restrict__ m1, float* __restrict__ m2,
                                    float* __restrict__ m3) {
  int i = blockIdx.x * 256 + threadIdx.x;  // point index over B*N_PTS
  if (i >= 8 * 16384) return;
  int b = i >> 14;
  float x = coords[2 * i], y = coords[2 * i + 1];
  const int Wd[3] = {52, 26, 13};
  float* mp[3];
  mp[0] = m1; mp[1] = m2; mp[2] = m3;
#pragma unroll
  for (int s = 0; s < 3; ++s) {
    int W = Wd[s];
    int xi = (int)floorf(x * ((float)W / 416.0f));
    int yi = (int)floorf(y * ((float)W / 416.0f));
    if (xi >= 0 && xi < W && yi >= 0 && yi < W)
      mp[s][(size_t)b * W * W + yi * W + xi] = 1.0f;
  }
}

__global__ void lidar_proj_kernel(const float* __restrict__ wlr, const float* __restrict__ lidar,
                                  float* __restrict__ v) {
  int b = blockIdx.x, o = threadIdx.x;
  const float* wr = wlr + (size_t)o * 1024;
  const float* ld = lidar + (size_t)b * 1024;
  float s = 0.f;
  for (int c = 0; c < 1024; ++c) s += wr[c] * ld[c];
  v[b * 256 + o] = s;
}

__global__ void u1c1_kernel(const float* __restrict__ wa1, const float* __restrict__ v,
                            const float* __restrict__ blr, float* __restrict__ u1,
                            float* __restrict__ c1) {
  int b = blockIdx.x, o = threadIdx.x;
  const float* wr = wa1 + (size_t)o * 512 + 256;
  const float* src = (b < 8) ? (v + b * 256) : blr;
  float s = 0.f;
  for (int m = 0; m < 256; ++m) s += wr[m] * src[m];
  if (b < 8) u1[b * 256 + o] = s; else c1[o] = s;
}

// ---------------------------------------------------------------------------
// Generic WMMA GEMM: out[b] = A16(MxK) @ B[b](KxN) + bias, block tile 128x64.
// Double-buffered LDS; async A prefetch overlaps with WMMA of current tile.
// EPI 0: store f16.  EPI 1: += resid[b], store f32.
// ---------------------------------------------------------------------------
template <typename TB, int EPI>
__global__ __launch_bounds__(128) void gemm_wmma_kernel(
    const _Float16* __restrict__ A16, const TB* __restrict__ Bsrc,
    const float* __restrict__ bias, const float* __restrict__ resid,
    _Float16* __restrict__ outH, float* __restrict__ outF, int M, int N, int K) {
  __shared__ __align__(16) _Float16 As[2][128 * 32];  // row-major [m][k]
  __shared__ __align__(16) _Float16 Bs[2][64 * 32];   // column-major [n][k]
  int tid = threadIdx.x, lane = tid & 31, wid = tid >> 5;
  int n0 = blockIdx.x * 64, m0 = blockIdx.y * 128, b = blockIdx.z;
  const TB* Bb = Bsrc + (size_t)b * K * N;

  auto stageA = [&](int kk, int buf) {
#pragma unroll
    for (int ii = 0; ii < 4; ++ii) {
      int i = tid + ii * 128;
      int r = i >> 2, q = i & 3;
      async_copy16(A16 + (size_t)(m0 + r) * K + kk + q * 8, &As[buf][r * 32 + q * 8]);
    }
  };
  auto stageB = [&](int kk, int buf) {
#pragma unroll
    for (int ii = 0; ii < 2; ++ii) {
      int i = tid + ii * 128;
      int n = i >> 2, ko = (i & 3) * 8;
      int col = n0 + n;
      alignas(16) _Float16 tmp[8];
#pragma unroll
      for (int t = 0; t < 8; ++t)
        tmp[t] = (col < N) ? (_Float16)(float)Bb[(size_t)(kk + ko + t) * N + col] : (_Float16)0.0f;
      *(uint4*)&Bs[buf][n * 32 + ko] = *(const uint4*)tmp;
    }
  };

  v8f acc[2][4] = {};
  stageA(0, 0);
  stageB(0, 0);
  async_wait0();
  __syncthreads();
  int T = K / 32;
  for (int t = 0; t < T; ++t) {
    int cur = t & 1;
    if (t + 1 < T) {
      stageA((t + 1) * 32, cur ^ 1);
      stageB((t + 1) * 32, cur ^ 1);
    }
#pragma unroll
    for (int mt = 0; mt < 2; ++mt) {
      v16h a = load_a_frag_rm(As[cur], mt * 64 + wid * 16, lane);
#pragma unroll
      for (int nt = 0; nt < 4; ++nt) {
        v16h bf = load_b_frag_cm(Bs[cur], nt * 16, 0, 32, lane);
        acc[mt][nt] = wmma16(a, bf, acc[mt][nt]);
      }
    }
    async_wait0();
    __syncthreads();
  }
  int h = lane >> 4;
  size_t outBase = (size_t)b * M * N;
#pragma unroll
  for (int mt = 0; mt < 2; ++mt)
#pragma unroll
    for (int nt = 0; nt < 4; ++nt) {
      int col = n0 + nt * 16 + (lane & 15);
      if (col < N) {
#pragma unroll
        for (int j = 0; j < 8; ++j) {
          int m = m0 + mt * 64 + wid * 16 + h * 8 + j;
          float v = acc[mt][nt][j] + bias[m];
          if (EPI == 1) {
            size_t idx = outBase + (size_t)m * N + col;
            outF[idx] = v + resid[idx];
          } else {
            outH[outBase + (size_t)m * N + col] = (_Float16)v;
          }
        }
      }
    }
}

// ---------------------------------------------------------------------------
// Attention + fusion: a = relu(W_a1L@img_r + mask*u1 + c1 + b_a1);
// att = softmax2(W_a2@a + b_a2); fused = img_r*att0 + lid_r*att1 (f16 out).
// Full M=256 x 32-column tile; img_r block resident column-major in LDS.
// ---------------------------------------------------------------------------
__global__ __launch_bounds__(256) void attn_fuse_kernel(
    const _Float16* __restrict__ imgR, const _Float16* __restrict__ Wa116,
    const float* __restrict__ ba1, const float* __restrict__ wa2,
    const float* __restrict__ ba2, const float* __restrict__ vvec,
    const float* __restrict__ u1, const float* __restrict__ c1,
    const float* __restrict__ blr, const float* __restrict__ mask,
    _Float16* __restrict__ fusedH, int N) {
  __shared__ __align__(16) _Float16 Bs[32 * 256];   // img_r block, column-major [n][k]
  __shared__ __align__(16) _Float16 As[256 * 32];   // W_a1 left-half K-tile, row-major
  __shared__ __align__(16) _Float16 aSh[256 * 32];  // activations [m][c]
  __shared__ float vS[256], u1S[256], c1S[256], blrS[256], ba1S[256];
  __shared__ float mS[32], att0S[32], att1S[32];
  int tid = threadIdx.x, lane = tid & 31, wid = tid >> 5;
  int n0 = blockIdx.x * 32, b = blockIdx.z;
  const _Float16* imgRb = imgR + (size_t)b * 256 * N;

  vS[tid] = vvec[b * 256 + tid]; u1S[tid] = u1[b * 256 + tid];
  c1S[tid] = c1[tid]; blrS[tid] = blr[tid]; ba1S[tid] = ba1[tid];
  if (tid < 32) {
    int col = n0 + tid;
    mS[tid] = (col < N) ? mask[(size_t)b * N + col] : 0.0f;
  }
  // stage img_r block column-major: 32 cols x 256 K, packed 8 at a time
#pragma unroll
  for (int ii = 0; ii < 4; ++ii) {
    int i = tid + ii * 256;
    int n = i >> 5, ko = (i & 31) * 8;
    int col = n0 + n;
    alignas(16) _Float16 tmp[8];
#pragma unroll
    for (int t = 0; t < 8; ++t)
      tmp[t] = (col < N) ? imgRb[(size_t)(ko + t) * N + col] : (_Float16)0.0f;
    *(uint4*)&Bs[n * 256 + ko] = *(const uint4*)tmp;
  }
  __syncthreads();

  v8f acc[2][2] = {};
  for (int kk = 0; kk < 256; kk += 32) {
    // async stage W_a1 left-half tile: 256 rows x 32 f16 = 1024 x 16B chunks
#pragma unroll
    for (int ii = 0; ii < 4; ++ii) {
      int i = tid + ii * 256;
      int r = i >> 2, q = i & 3;
      async_copy16(Wa116 + (size_t)r * 512 + kk + q * 8, &As[r * 32 + q * 8]);
    }
    async_wait0();
    __syncthreads();
#pragma unroll
    for (int mt = 0; mt < 2; ++mt) {
      v16h a = load_a_frag_rm(As, wid * 32 + mt * 16, lane);
#pragma unroll
      for (int nt = 0; nt < 2; ++nt) {
        v16h bf = load_b_frag_cm(Bs, nt * 16, kk, 256, lane);
        acc[mt][nt] = wmma16(a, bf, acc[mt][nt]);
      }
    }
    __syncthreads();
  }

  int h = lane >> 4;
#pragma unroll
  for (int mt = 0; mt < 2; ++mt)
#pragma unroll
    for (int nt = 0; nt < 2; ++nt) {
      int c = nt * 16 + (lane & 15);
#pragma unroll
      for (int j = 0; j < 8; ++j) {
        int m = wid * 32 + mt * 16 + h * 8 + j;
        float val = acc[mt][nt][j] + mS[c] * u1S[m] + c1S[m] + ba1S[m];
        aSh[m * 32 + c] = (_Float16)fmaxf(val, 0.0f);
      }
    }
  __syncthreads();

  if (tid < 32) {
    float z0 = ba2[0], z1 = ba2[1];
    for (int m = 0; m < 256; ++m) {
      float am = (float)aSh[m * 32 + tid];
      z0 += wa2[m] * am;
      z1 += wa2[256 + m] * am;
    }
    float mx = fmaxf(z0, z1);
    float e0 = __expf(z0 - mx), e1 = __expf(z1 - mx);
    float inv = 1.0f / (e0 + e1);
    att0S[tid] = e0 * inv; att1S[tid] = e1 * inv;
  }
  __syncthreads();

  int m = tid;
  for (int c = 0; c < 32; ++c) {
    int col = n0 + c;
    if (col < N) {
      float ir = (float)Bs[c * 256 + m];  // img_r from column-major block
      float lr = mS[c] * vS[m] + blrS[m];
      float f = ir * att0S[c] + lr * att1S[c];
      fusedH[(size_t)b * 256 * N + (size_t)m * N + col] = (_Float16)f;
    }
  }
}

// ---------------------------------------------------------------------------
// Implicit-GEMM 3x3 conv (SAME) + folded BN + ReLU. M=256, K=2304, N=H*W.
// Double-buffered; B gathered with halo/zero-pad into column-major LDS.
// ---------------------------------------------------------------------------
__global__ __launch_bounds__(128) void conv3x3_wmma_kernel(
    const _Float16* __restrict__ Wf16, const _Float16* __restrict__ fusedH,
    const float* __restrict__ bf, const float* __restrict__ gamma,
    const float* __restrict__ beta, const float* __restrict__ mean,
    const float* __restrict__ var, _Float16* __restrict__ outH, int Wd, int Hd) {
  const int K = 2304;
  int N = Wd * Hd;
  __shared__ __align__(16) _Float16 As[2][128 * 32];
  __shared__ __align__(16) _Float16 Bs[2][64 * 32];  // column-major [n][k]
  int tid = threadIdx.x, lane = tid & 31, wid = tid >> 5;
  int n0 = blockIdx.x * 64, m0 = blockIdx.y * 128, b = blockIdx.z;
  const _Float16* fb = fusedH + (size_t)b * 256 * N;

  auto stageA = [&](int kk, int buf) {
#pragma unroll
    for (int ii = 0; ii < 4; ++ii) {
      int i = tid + ii * 128;
      int r = i >> 2, q = i & 3;
      async_copy16(Wf16 + (size_t)(m0 + r) * K + kk + q * 8, &As[buf][r * 32 + q * 8]);
    }
  };
  auto stageB = [&](int kk, int buf) {
#pragma unroll
    for (int ii = 0; ii < 2; ++ii) {
      int i = tid + ii * 128;
      int n = i >> 2, ko = (i & 3) * 8;
      int col = n0 + n;
      int y = col / Wd, x = col - y * Wd;
      alignas(16) _Float16 tmp[8];
#pragma unroll
      for (int t = 0; t < 8; ++t) {
        int k = kk + ko + t;
        int ch = k / 9, tt = k - ch * 9;
        int dy = tt / 3 - 1, dx = tt - (tt / 3) * 3 - 1;
        int yy = y + dy, xx = x + dx;
        _Float16 val = (_Float16)0.0f;
        if (col < N && yy >= 0 && yy < Hd && xx >= 0 && xx < Wd)
          val = fb[(size_t)ch * N + yy * Wd + xx];
        tmp[t] = val;
      }
      *(uint4*)&Bs[buf][n * 32 + ko] = *(const uint4*)tmp;
    }
  };

  v8f acc[2][4] = {};
  stageA(0, 0);
  stageB(0, 0);
  async_wait0();
  __syncthreads();
  const int T = K / 32;
  for (int t = 0; t < T; ++t) {
    int cur = t & 1;
    if (t + 1 < T) {
      stageA((t + 1) * 32, cur ^ 1);
      stageB((t + 1) * 32, cur ^ 1);
    }
#pragma unroll
    for (int mt = 0; mt < 2; ++mt) {
      v16h a = load_a_frag_rm(As[cur], mt * 64 + wid * 16, lane);
#pragma unroll
      for (int nt = 0; nt < 4; ++nt) {
        v16h bfr = load_b_frag_cm(Bs[cur], nt * 16, 0, 32, lane);
        acc[mt][nt] = wmma16(a, bfr, acc[mt][nt]);
      }
    }
    async_wait0();
    __syncthreads();
  }
  int h = lane >> 4;
  size_t outBase = (size_t)b * 256 * N;
#pragma unroll
  for (int mt = 0; mt < 2; ++mt)
#pragma unroll
    for (int nt = 0; nt < 4; ++nt) {
      int col = n0 + nt * 16 + (lane & 15);
      if (col < N) {
#pragma unroll
        for (int j = 0; j < 8; ++j) {
          int m = m0 + mt * 64 + wid * 16 + h * 8 + j;
          float s = gamma[m] * rsqrtf(var[m] + 1e-5f);
          float sh = beta[m] - mean[m] * s;
          float v = (acc[mt][nt][j] + bf[m]) * s + sh;
          outH[outBase + (size_t)m * N + col] = (_Float16)fmaxf(v, 0.0f);
        }
      }
    }
}

// ---------------------------------------------------------------------------
extern "C" void kernel_launch(void* const* d_in, const int* in_sizes, int n_in,
                              void* d_out, int out_size, void* d_ws, size_t ws_size,
                              hipStream_t stream) {
  (void)in_sizes; (void)n_in; (void)out_size; (void)ws_size;
  const float* img[3] = {(const float*)d_in[0], (const float*)d_in[1], (const float*)d_in[2]};
  const float* lidar  = (const float*)d_in[3];
  const float* coords = (const float*)d_in[4];
  const int Cimg[3] = {256, 512, 1024};
  const int Wd[3]   = {52, 26, 13};
  auto P = [&](int s, int j) { return (const float*)d_in[5 + s * 16 + j]; };

  char* w = (char*)d_ws;
  auto alloc = [&](size_t bytes) {
    char* p = w;
    w += (bytes + 255) & ~(size_t)255;
    return p;
  };
  float* mask[3];
  for (int s = 0; s < 3; ++s) mask[s] = (float*)alloc((size_t)8 * Wd[s] * Wd[s] * 4);
  float* vbuf  = (float*)alloc(8 * 256 * 4);
  float* u1buf = (float*)alloc(8 * 256 * 4);
  float* c1buf = (float*)alloc(256 * 4);
  _Float16* wir16 = (_Float16*)alloc((size_t)256 * 1024 * 2);
  _Float16* wa116 = (_Float16*)alloc((size_t)256 * 512 * 2);
  _Float16* wf16  = (_Float16*)alloc((size_t)256 * 2304 * 2);
  _Float16* wo16  = (_Float16*)alloc((size_t)1024 * 256 * 2);
  _Float16* bufA  = (_Float16*)alloc((size_t)8 * 256 * 2704 * 2);  // img_r, later f
  _Float16* bufB  = (_Float16*)alloc((size_t)8 * 256 * 2704 * 2);  // fused

  for (int s = 0; s < 3; ++s) {
    int n = 8 * Wd[s] * Wd[s];
    zero_f32_kernel<<<(n + 255) / 256, 256, 0, stream>>>(mask[s], n);
  }
  scatter_mask_kernel<<<(8 * 16384 + 255) / 256, 256, 0, stream>>>(coords, mask[0], mask[1], mask[2]);

  size_t outOff = 0;
  for (int s = 0; s < 3; ++s) {
    int C = Cimg[s], W_ = Wd[s], N = W_ * W_;
    auto cvt = [&](const float* src, _Float16* dst, int n) {
      f32_to_f16_kernel<<<(n + 255) / 256, 256, 0, stream>>>(src, dst, n);
    };
    cvt(P(s, 0), wir16, 256 * C);
    cvt(P(s, 4), wa116, 256 * 512);
    cvt(P(s, 8), wf16, 256 * 2304);
    cvt(P(s, 14), wo16, C * 256);
    lidar_proj_kernel<<<8, 256, 0, stream>>>(P(s, 2), lidar, vbuf);
    u1c1_kernel<<<9, 256, 0, stream>>>(P(s, 4), vbuf, P(s, 3), u1buf, c1buf);

    // img_r = W_ir @ img + b_ir  (f16)
    dim3 g1((N + 63) / 64, 256 / 128, 8);
    gemm_wmma_kernel<float, 0><<<g1, 128, 0, stream>>>(
        wir16, img[s], P(s, 1), nullptr, bufA, nullptr, 256, N, C);

    // attention + blend -> fused (f16)
    dim3 g2((N + 31) / 32, 1, 8);
    attn_fuse_kernel<<<g2, 256, 0, stream>>>(
        bufA, wa116, P(s, 5), P(s, 6), P(s, 7), vbuf, u1buf, c1buf, P(s, 3),
        mask[s], bufB, N);

    // f = relu(BN(conv3x3(fused) + b_f))  (f16, overwrites img_r buffer)
    dim3 g3((N + 63) / 64, 2, 8);
    conv3x3_wmma_kernel<<<g3, 128, 0, stream>>>(
        wf16, bufB, P(s, 9), P(s, 10), P(s, 11), P(s, 12), P(s, 13), bufA, W_, W_);

    // out = W_o @ f + b_o + img  (f32)
    dim3 g4((N + 63) / 64, C / 128, 8);
    gemm_wmma_kernel<_Float16, 1><<<g4, 128, 0, stream>>>(
        wo16, bufA, P(s, 15), img[s], nullptr, (float*)d_out + outOff, C, N, 256);

    outOff += (size_t)8 * C * N;
  }
}
